// MultiHeadAttention_42167988912698
// MI455X (gfx1250) — matrix operand order repaired and checked by reading, not compiled, not run on hardware
//
#include <hip/hip_runtime.h>
#include <hip/hip_bf16.h>

typedef __bf16 bf16_t;
typedef __attribute__((ext_vector_type(16))) __bf16 v16bf;
typedef __attribute__((ext_vector_type(8)))  __bf16 v8bf;
typedef __attribute__((ext_vector_type(8)))  float  v8f;
typedef __attribute__((ext_vector_type(4)))  float  v4f;

#define BB  4
#define SS  2048
#define DD  1024
#define HH  16
#define HDD 64

// ---------------------------------------------------------------------------
// WMMA helpers (CDNA5 wave32, v_wmma_f32_16x16x32_bf16)
// ---------------------------------------------------------------------------
static __device__ __forceinline__ v8f wmma_bf16(v16bf a, v16bf b, v8f c) {
  // (neg_a, A, neg_b, B, c_mod, C, reuse_a, reuse_b)
  return __builtin_amdgcn_wmma_f32_16x16x32_bf16(false, a, false, b,
                                                 (short)0, c, false, false);
}

// A fragment (16x32, M x K): lane m=l&15, half h=l>>4 holds K = 16h+{0..15}
static __device__ __forceinline__ v16bf load_a_bf16(
    const bf16_t* __restrict__ base, int stride, int lane) {
  const int r = lane & 15, h = lane >> 4;
  const bf16_t* p = base + (size_t)r * stride + 16 * h;
  v8bf lo = *(const v8bf*)(p);
  v8bf hi = *(const v8bf*)(p + 8);
  v16bf f;
#pragma unroll
  for (int j = 0; j < 8; ++j) { f[j] = lo[j]; f[8 + j] = hi[j]; }
  return f;
}

// B fragment (32x16, K x N), source stored row-major as [N rows][K contiguous]
// lane n=l&15, half h=l>>4 holds K = 16h + {0..15} sequential -> one 32B load
static __device__ __forceinline__ v16bf load_b_bf16(
    const bf16_t* __restrict__ base, int stride, int lane) {
  const int n = lane & 15, h = lane >> 4;
  return *(const v16bf*)(base + (size_t)n * stride + 16 * h);
}

// ---------------------------------------------------------------------------
// fp32 -> bf16 conversion (vectorized 4-wide); used for weights and q/k/v
// ---------------------------------------------------------------------------
__global__ void f32_to_bf16_kernel(const float* __restrict__ in,
                                   bf16_t* __restrict__ out, int n4) {
  int i = blockIdx.x * blockDim.x + threadIdx.x;
  if (i < n4) {
    v4f x = *(const v4f*)(in + 4 * i);
#pragma unroll
    for (int j = 0; j < 4; ++j) out[4 * i + j] = (bf16_t)x[j];
  }
}

// ---------------------------------------------------------------------------
// GEMM: Y = X[M=8192,1024](bf16) @ W[1024,1024]^T(bf16) + bias
// Each wave computes a 32(M) x 64(N) tile: 2 A-frags x 4 B-frags = 8 WMMA
// per 32-K step; B fragments reused across both M halves. Zero conversion
// VALU in the K-loop (all operands pre-converted to bf16).
// MODE 0: write bf16 head-split [B,H,S,64]   (Q, K)
// MODE 1: write bf16 transposed [B,H,64,S]   (V)
// MODE 2: write fp32 plain [M,1024]          (output projection -> d_out)
// ---------------------------------------------------------------------------
template <int MODE>
__global__ __launch_bounds__(256) void gemm_bf16_kernel(
    const bf16_t* __restrict__ X, const bf16_t* __restrict__ W,
    const float* __restrict__ bias, void* __restrict__ outp) {
  const int lane = threadIdx.x & 31;
  const int wave = blockIdx.x * (blockDim.x >> 5) + (threadIdx.x >> 5);
  const int NT = DD / 64;                  // 16 N-tiles of 64
  const int m0 = (wave / NT) * 32;
  const int n0 = (wave % NT) * 64;

  v8f acc[2][4];
  v8f vz = {};
#pragma unroll
  for (int mi = 0; mi < 2; ++mi)
#pragma unroll
    for (int st = 0; st < 4; ++st) acc[mi][st] = vz;

  for (int k0 = 0; k0 < DD; k0 += 32) {
    // prefetch next X chunk (global_prefetch_b8)
    if (k0 + 32 < DD)
      __builtin_prefetch(X + (size_t)(m0 + (lane & 15)) * DD + k0 + 32, 0, 3);
    v16bf a0 = load_a_bf16(X + (size_t)m0 * DD + k0, DD, lane);
    v16bf a1 = load_a_bf16(X + (size_t)(m0 + 16) * DD + k0, DD, lane);
#pragma unroll
    for (int st = 0; st < 4; ++st) {
      v16bf b = load_b_bf16(W + (size_t)(n0 + st * 16) * DD + k0, DD, lane);
      acc[0][st] = wmma_bf16(a0, b, acc[0][st]);
      acc[1][st] = wmma_bf16(a1, b, acc[1][st]);
    }
  }

  const int nl = lane & 15, h = lane >> 4;
#pragma unroll
  for (int st = 0; st < 4; ++st) {
    const int n = n0 + st * 16 + nl;
    const float bv = bias[n];
    const int hgl = n >> 6, hd = n & 63;
#pragma unroll
    for (int mi = 0; mi < 2; ++mi) {
#pragma unroll
      for (int r = 0; r < 8; ++r) {
        const int m = m0 + mi * 16 + h * 8 + r;
        const int bi = m >> 11, s = m & (SS - 1);
        const float val = acc[mi][st][r] + bv;
        if (MODE == 0) {
          ((bf16_t*)outp)[(((size_t)(bi * HH + hgl)) * SS + s) * HDD + hd] =
              (bf16_t)val;
        } else if (MODE == 1) {
          ((bf16_t*)outp)[(((size_t)(bi * HH + hgl)) * HDD + hd) * SS + s] =
              (bf16_t)val;
        } else {
          ((float*)outp)[(size_t)m * DD + n] = val;
        }
      }
    }
  }
}

// ---------------------------------------------------------------------------
// One 32-key flash-attention block. MASKED=true only for the diagonal block.
// ---------------------------------------------------------------------------
template <bool MASKED>
static __device__ __forceinline__ void attn_block(
    int s0, int q0, int lane, const bf16_t* __restrict__ kh_bh,
    const bf16_t* __restrict__ vT_bh, const v16bf& aq0, const v16bf& aq1,
    float* __restrict__ row_m, float* __restrict__ row_l,
    v8f* __restrict__ acc_o, bf16_t (*__restrict__ Pl)[32]) {
  const int nl = lane & 15, h = lane >> 4;
  v8f vz = {};
  float P[2][8];
  float tmax[8];
#pragma unroll
  for (int r = 0; r < 8; ++r) tmax[r] = -3.0e38f;

  // prefetch next key block (global_prefetch_b8)
  __builtin_prefetch(kh_bh + (size_t)(s0 + 32 + nl) * HDD, 0, 3);

  // ---- logits: two 16x16 tiles (keys s0..s0+31), QK^T over HD=64 ----
#pragma unroll
  for (int st = 0; st < 2; ++st) {
    const int ks = s0 + st * 16;
    const bf16_t* kbase = kh_bh + (size_t)ks * HDD;
    v16bf bk0 = load_b_bf16(kbase, HDD, lane);
    v16bf bk1 = load_b_bf16(kbase + 32, HDD, lane);
    v8f accs = vz;
    accs = wmma_bf16(aq0, bk0, accs);
    accs = wmma_bf16(aq1, bk1, accs);
#pragma unroll
    for (int r = 0; r < 8; ++r) {
      float lg = accs[r] * 0.125f;  // 1/sqrt(64)
      if (MASKED) {
        const int qg = q0 + h * 8 + r;
        const int kg = ks + nl;
        if (kg > qg) lg = -1.0e30f;  // causal mask (diagonal block only)
      }
      P[st][r] = lg;
      tmax[r] = fmaxf(tmax[r], lg);
    }
  }

  // ---- online softmax: row reductions across the 16 lanes of each half ----
#pragma unroll
  for (int r = 0; r < 8; ++r) {
    float v = tmax[r];
    v = fmaxf(v, __shfl_xor(v, 1, 32));
    v = fmaxf(v, __shfl_xor(v, 2, 32));
    v = fmaxf(v, __shfl_xor(v, 4, 32));
    v = fmaxf(v, __shfl_xor(v, 8, 32));
    const float m_new = fmaxf(row_m[r], v);
    const float corr = __expf(row_m[r] - m_new);
    row_m[r] = m_new;
    float rs = 0.0f;
#pragma unroll
    for (int st = 0; st < 2; ++st) {
      const float p = __expf(P[st][r] - m_new);
      P[st][r] = p;
      rs += p;
    }
    rs += __shfl_xor(rs, 1, 32);
    rs += __shfl_xor(rs, 2, 32);
    rs += __shfl_xor(rs, 4, 32);
    rs += __shfl_xor(rs, 8, 32);
    row_l[r] = row_l[r] * corr + rs;
#pragma unroll
    for (int dt = 0; dt < 4; ++dt) acc_o[dt][r] *= corr;
    // stage P (C/D layout) into LDS as bf16 for A-fragment reload
#pragma unroll
    for (int st = 0; st < 2; ++st)
      Pl[h * 8 + r][st * 16 + nl] = (bf16_t)P[st][r];
  }

  __builtin_amdgcn_wave_barrier();
  asm volatile("s_wait_dscnt 0" ::: "memory");

  // ---- PV: A = P (16x32), B = vT rows (32 s x 16 d per subtile) ----
  const v16bf ap = load_a_bf16(&Pl[0][0], 32, lane);
#pragma unroll
  for (int dt = 0; dt < 4; ++dt) {
    const bf16_t* vbase = vT_bh + (size_t)(dt * 16) * SS + s0;
    v16bf bv = load_b_bf16(vbase, SS, lane);
    acc_o[dt] = wmma_bf16(ap, bv, acc_o[dt]);
  }
}

// ---------------------------------------------------------------------------
// Fused causal flash attention. One wave per 16-query tile of one (b,h).
// qh/kh: bf16 [B,H,S,64]; vT: bf16 [B,H,64,S]
// out: bf16 [B,S,D] (head-concat), ready as A-operand of the final GEMM
// ---------------------------------------------------------------------------
__global__ __launch_bounds__(128) void attn_fused_kernel(
    const bf16_t* __restrict__ qh, const bf16_t* __restrict__ kh,
    const bf16_t* __restrict__ vT, bf16_t* __restrict__ attn_out) {
  __shared__ bf16_t Pl[4][16][32];  // per-wave P staging (C-layout -> A-frag)

  const int lane = threadIdx.x & 31;
  const int wv = threadIdx.x >> 5;
  const int wave = blockIdx.x * 4 + wv;
  const int QT = SS / 16;                  // 128 query tiles
  const int bh = wave / QT;
  const int q0 = (wave % QT) * 16;
  const int nl = lane & 15, h = lane >> 4;

  const bf16_t* kh_bh = kh + (size_t)bh * SS * HDD;
  const bf16_t* vT_bh = vT + (size_t)bh * HDD * SS;

  // Q tile A-fragments (16 x 64 = two k-steps), loaded once
  const bf16_t* qbase = qh + ((size_t)bh * SS + q0) * HDD;
  const v16bf aq0 = load_a_bf16(qbase, HDD, lane);
  const v16bf aq1 = load_a_bf16(qbase + 32, HDD, lane);

  float row_m[8], row_l[8];
  v8f acc_o[4];
  v8f vz = {};
#pragma unroll
  for (int r = 0; r < 8; ++r) { row_m[r] = -3.0e38f; row_l[r] = 0.0f; }
#pragma unroll
  for (int dt = 0; dt < 4; ++dt) acc_o[dt] = vz;

  // Unmasked full blocks: keys [0, nfull*32) are strictly below the diagonal
  // for every row in this 16-query tile (s0+31 <= q0).
  const int nfull = q0 >> 5;
  for (int jb = 0; jb < nfull; ++jb)
    attn_block<false>(jb * 32, q0, lane, kh_bh, vT_bh, aq0, aq1, row_m, row_l,
                      acc_o, Pl[wv]);
  // Single diagonal block with causal masking.
  attn_block<true>(nfull * 32, q0, lane, kh_bh, vT_bh, aq0, aq1, row_m, row_l,
                   acc_o, Pl[wv]);

  // ---- epilogue: O /= l, write head-concat bf16 [B,S,D] ----
  const int b = bh >> 4, hh = bh & 15;
#pragma unroll
  for (int r = 0; r < 8; ++r) {
    const float inv_l = 1.0f / row_l[r];
    const int q = q0 + h * 8 + r;
#pragma unroll
    for (int dt = 0; dt < 4; ++dt) {
      const int d = hh * HDD + dt * 16 + nl;
      attn_out[((size_t)b * SS + q) * DD + d] =
          (bf16_t)(acc_o[dt][r] * inv_l);
    }
  }
}

// ---------------------------------------------------------------------------
// Host launcher
// ---------------------------------------------------------------------------
extern "C" void kernel_launch(void* const* d_in, const int* in_sizes, int n_in,
                              void* d_out, int out_size, void* d_ws,
                              size_t ws_size, hipStream_t stream) {
  (void)in_sizes; (void)n_in; (void)out_size; (void)ws_size;
  const float* q  = (const float*)d_in[0];
  const float* k  = (const float*)d_in[1];
  const float* v  = (const float*)d_in[2];
  // d_in[3] = mask (causal; applied analytically in-kernel)
  const float* Wq = (const float*)d_in[4];
  const float* bq = (const float*)d_in[5];
  const float* Wk = (const float*)d_in[6];
  const float* bk = (const float*)d_in[7];
  const float* Wv = (const float*)d_in[8];
  const float* bv = (const float*)d_in[9];
  const float* Wo = (const float*)d_in[10];
  const float* bo = (const float*)d_in[11];

  char* ws = (char*)d_ws;
  size_t off = 0;
  const size_t WBYTES = (size_t)DD * DD * sizeof(bf16_t);     // 2 MB
  const size_t XBYTES = (size_t)BB * SS * DD * sizeof(bf16_t); // 16 MB
  bf16_t* Wqb = (bf16_t*)(ws + off); off += WBYTES;
  bf16_t* Wkb = (bf16_t*)(ws + off); off += WBYTES;
  bf16_t* Wvb = (bf16_t*)(ws + off); off += WBYTES;
  bf16_t* Wob = (bf16_t*)(ws + off); off += WBYTES;
  bf16_t* qb  = (bf16_t*)(ws + off); off += XBYTES;  // q in bf16
  bf16_t* kb  = (bf16_t*)(ws + off); off += XBYTES;  // k in bf16
  bf16_t* vb  = (bf16_t*)(ws + off); off += XBYTES;  // v in bf16
  bf16_t* qhb = (bf16_t*)(ws + off); off += XBYTES;  // Q heads [B,H,S,64]
  bf16_t* khb = (bf16_t*)(ws + off); off += XBYTES;  // K heads [B,H,S,64]
  bf16_t* vTb = (bf16_t*)(ws + off); off += XBYTES;  // V heads T [B,H,64,S]
  bf16_t* att = (bf16_t*)(ws + off); off += XBYTES;  // attention out [B,S,D]

  const int WN4 = DD * DD / 4;                 // weight quads
  const int XN4 = BB * SS * DD / 4;            // activation quads
  f32_to_bf16_kernel<<<(WN4 + 255) / 256, 256, 0, stream>>>(Wq, Wqb, WN4);
  f32_to_bf16_kernel<<<(WN4 + 255) / 256, 256, 0, stream>>>(Wk, Wkb, WN4);
  f32_to_bf16_kernel<<<(WN4 + 255) / 256, 256, 0, stream>>>(Wv, Wvb, WN4);
  f32_to_bf16_kernel<<<(WN4 + 255) / 256, 256, 0, stream>>>(Wo, Wob, WN4);
  f32_to_bf16_kernel<<<(XN4 + 255) / 256, 256, 0, stream>>>(q, qb, XN4);
  f32_to_bf16_kernel<<<(XN4 + 255) / 256, 256, 0, stream>>>(k, kb, XN4);
  f32_to_bf16_kernel<<<(XN4 + 255) / 256, 256, 0, stream>>>(v, vb, XN4);

  // 8192/32 M-tiles * 16 N-tiles = 4096 wave-tiles, 8 waves per 256-thr block
  const int gemm_blocks = (BB * SS / 32) * (DD / 64) / 8;  // 512
  gemm_bf16_kernel<0><<<gemm_blocks, 256, 0, stream>>>(qb, Wqb, bq, qhb);
  gemm_bf16_kernel<0><<<gemm_blocks, 256, 0, stream>>>(kb, Wkb, bk, khb);
  gemm_bf16_kernel<1><<<gemm_blocks, 256, 0, stream>>>(vb, Wvb, bv, vTb);

  // B*H * S/16 = 8192 waves, 4 waves per 128-thr block
  attn_fused_kernel<<<BB * HH * (SS / 16) / 4, 128, 0, stream>>>(qhb, khb, vTb,
                                                                 att);

  gemm_bf16_kernel<2><<<gemm_blocks, 256, 0, stream>>>(att, Wob, bo, d_out);
}
